// BACPI_42408507081161
// MI455X (gfx1250) — compile-verified, matrix operand order checked by reading, not run on hardware
//
#include <hip/hip_runtime.h>
#include <cstdint>

typedef __attribute__((ext_vector_type(2)))  float  v2f;
typedef __attribute__((ext_vector_type(8)))  float  v8f;
typedef __attribute__((ext_vector_type(16))) __bf16 v16bf;

// ---------- helpers ----------
static __device__ __forceinline__ unsigned f2bf1(float f) {
    unsigned u = __builtin_bit_cast(unsigned, f);
    return (u + 0x7FFFu + ((u >> 16) & 1u)) >> 16;   // RNE f32 -> bf16
}
static __device__ __forceinline__ unsigned pack2bf(float lo, float hi) {
    return f2bf1(lo) | (f2bf1(hi) << 16);
}

// =====================================================================
// Kernel 1/3: Wh = x @ W  (f32 WMMA 16x16x4), emit:
//   * Vfrag: bf16 B-fragment layout for v_wmma_f32_16x16x32_bf16
//            [jc(16)][nt(NTV)][lane(32)][r(8)] u32  (per image)
//   * e1/e2 folded into the GEMM: two extra W columns w1 = W@a1, w2 = W@a2
//     (e1 = Wh@a1 = x@(W@a1)), read out of the extra N-tile's accumulator.
// W staged pair-interleaved in LDS: Wlp[k/2][2n+par] = W[k][n], so each
// B operand is a single aligned ds_load_b64 (no register glue).
// grid: (2, heads, 32 batches), block 256 (8 waves, each wave: 32 rows)
// =====================================================================
template<int KSTEPS, int KLIM, int NTV, int FO>
__global__ __launch_bounds__(256)
void gat_gemm(const float* __restrict__ x, int xstride,
              const float* __restrict__ Wg, const float* __restrict__ avec,
              unsigned* __restrict__ vfrag,
              float* __restrict__ e1, float* __restrict__ e2)
{
    constexpr int NP = (NTV + 1) * 16;                 // value cols + e-tile
    __shared__ __attribute__((aligned(16))) float Wlp[KSTEPS * 2][NP * 2];

    const int tid = threadIdx.x;
    const int img = blockIdx.y * gridDim.z + blockIdx.z;   // head*32 + batch
    x     += (size_t)blockIdx.z * 512 * xstride;
    Wg    += (size_t)blockIdx.y * KLIM * FO;
    avec  += (size_t)blockIdx.y * 2 * FO;
    vfrag += (size_t)img * 16 * NTV * 256;
    e1    += (size_t)img * 512;
    e2    += (size_t)img * 512;

    // phase 1: stage W pair-interleaved, zero-padded
    for (int idx = tid; idx < KSTEPS * 2 * NP; idx += 256) {
        const int kp = idx / NP, n = idx % NP;
#pragma unroll
        for (int par = 0; par < 2; ++par) {
            const int k = 2 * kp + par;
            Wlp[kp][2 * n + par] = (k < KLIM && n < FO) ? Wg[k * FO + n] : 0.0f;
        }
    }
    __syncthreads();
    // phase 2: e-columns  w1[k] = dot(W[k,:], a1), w2[k] = dot(W[k,:], a2)
    if (tid < KSTEPS * 4) {
        const int k = tid;
        float w1 = 0.0f, w2 = 0.0f;
        if (k < KLIM) {
            for (int c = 0; c < FO; ++c) {
                const float wv = Wg[k * FO + c];
                w1 += wv * avec[c];
                w2 += wv * avec[FO + c];
            }
        }
        Wlp[k >> 1][2 * (NTV * 16 + 0) + (k & 1)] = w1;
        Wlp[k >> 1][2 * (NTV * 16 + 1) + (k & 1)] = w2;
    }
    __syncthreads();

    const int wave = tid >> 5, lane = tid & 31;
    const int h = lane >> 4, ln = lane & 15;
    const int jc = blockIdx.x * 8 + wave;              // 32-row block, 0..15
    const float* xr0 = x + (size_t)(jc * 32 + ln) * xstride;
    const float* xr1 = xr0 + (size_t)16 * xstride;

    unsigned* vout = vfrag + (((size_t)jc * NTV) * 32 + lane) * 8;

#pragma unroll
    for (int nt = 0; nt <= NTV; ++nt) {
        v8f accA = {}; v8f accB = {};
        const int n = nt * 16 + ln;
#pragma unroll
        for (int ks = 0; ks < KSTEPS; ++ks) {
            const int k0 = ks * 4 + 2 * h;
            const v2f b = *(const v2f*)&Wlp[2 * ks + h][2 * n];
            v2f aA, aB;
            aA.x = (k0     < KLIM) ? xr0[k0]     : 0.0f;
            aA.y = (k0 + 1 < KLIM) ? xr0[k0 + 1] : 0.0f;
            aB.x = (k0     < KLIM) ? xr1[k0]     : 0.0f;
            aB.y = (k0 + 1 < KLIM) ? xr1[k0 + 1] : 0.0f;
            accA = __builtin_amdgcn_wmma_f32_16x16x4_f32(false, aA, false, b,
                                                         (short)0, accA, false, false);
            accB = __builtin_amdgcn_wmma_f32_16x16x4_f32(false, aB, false, b,
                                                         (short)0, accB, false, false);
        }
        if (nt < NTV) {
            // pack into bf16 B-fragment layout for 16x16x32 bf16 WMMA:
            //   r<4 : K = 2r+8h        -> accA rows 2r, 2r+1
            //   r>=4: K = 16+2(r-4)+8h -> accB rows 2(r-4), 2(r-4)+1
            union { unsigned u[8]; uint4 q[2]; } fr;
#pragma unroll
            for (int r = 0; r < 4; ++r)
                fr.u[r] = pack2bf(accA[2 * r], accA[2 * r + 1]);
#pragma unroll
            for (int r = 4; r < 8; ++r)
                fr.u[r] = pack2bf(accB[2 * (r - 4)], accB[2 * (r - 4) + 1]);
            uint4* dst = (uint4*)(vout + (size_t)nt * 32 * 8);
            dst[0] = fr.q[0]; dst[1] = fr.q[1];
        } else {
            // e-tile: col 0 = e1, col 1 = e2 (C layout rows r+8h / 16+r+8h)
            const int base = jc * 32;
            if (ln == 0) {
#pragma unroll
                for (int r = 0; r < 8; ++r) {
                    e1[base + 8 * h + r]      = accA[r];
                    e1[base + 16 + 8 * h + r] = accB[r];
                }
            } else if (ln == 1) {
#pragma unroll
                for (int r = 0; r < 8; ++r) {
                    e2[base + 8 * h + r]      = accA[r];
                    e2[base + 16 + 8 * h + r] = accB[r];
                }
            }
        }
    }
}

// =====================================================================
// Kernel 2/4: out = softmax_j(mask(leaky(e1_i+e2_j))) @ Wh  [+ ELU]
// No-max softmax (scores are O(±6), masked -> p = 0 exactly).
// P built directly in bf16 A-fragment layout; V staged into LDS via
// global_load_async_to_lds_b128 (two phases of 8 j-chunks), B-fragments
// read back with two ds_load_b128 per tile.
// grid: (8 rowchunks, heads, 32 batches), block 128 (4 waves x 16 rows)
// =====================================================================
template<int NT, int FO, int ELU>
__global__ __launch_bounds__(128)
void gat_attn(const unsigned* __restrict__ vfrag,
              const float* __restrict__ e1, const float* __restrict__ e2,
              const int* __restrict__ adj,
              float* __restrict__ out, int ostride, int headcolstride)
{
    __shared__ __attribute__((aligned(16))) unsigned vl[8 * NT * 256];

    const int tid = threadIdx.x, wave = tid >> 5, lane = tid & 31;
    const int h = lane >> 4, ln = lane & 15;
    const int bb = blockIdx.z;
    const int img = blockIdx.y * gridDim.z + bb;
    vfrag += (size_t)img * 16 * NT * 256;
    e1    += (size_t)img * 512;
    e2    += (size_t)img * 512;

    const int rowbase = blockIdx.x * 64 + wave * 16;
    const float e1i = e1[rowbase + ln];                 // A-layout: lane's row is l%16
    const int* arow = adj + ((size_t)bb * 512 + rowbase + ln) * 512;

    v8f acc[NT] = {};
    float rs = 0.0f;

    for (int phase = 0; phase < 2; ++phase) {
        // async-stage 8 j-chunks of V fragments (8*NT KB) into LDS
        const unsigned* src = vfrag + (size_t)phase * 8 * NT * 256;
        const int n16 = 8 * NT * 256 / 4;               // 16B chunks
        for (int ofs = tid; ofs < n16; ofs += 128) {
            unsigned la = (unsigned)(size_t)(vl + ofs * 4);
            const unsigned* g = src + ofs * 4;
            asm volatile("global_load_async_to_lds_b128 %0, %1, off"
                         :: "v"(la), "v"(g) : "memory");
        }
        asm volatile("s_wait_asynccnt 0" ::: "memory");
        __syncthreads();

#pragma unroll
        for (int jcl = 0; jcl < 8; ++jcl) {
            const int jc = phase * 8 + jcl;
            union { unsigned u[8]; v16bf v; } pf;       // P: bf16 A-fragment
#pragma unroll
            for (int r = 0; r < 8; ++r) {
                const int k = (r < 4) ? (2 * r + 8 * h) : (16 + 2 * (r - 4) + 8 * h);
                const int j = jc * 32 + k;
                const int2 ad = *(const int2*)(arow + j);
                const float2 ev = *(const float2*)(e2 + j);
                float s0 = e1i + ev.x;  s0 = (s0 > 0.0f) ? s0 : 0.2f * s0;
                float s1 = e1i + ev.y;  s1 = (s1 > 0.0f) ? s1 : 0.2f * s1;
                const float p0 = ad.x ? __expf(s0) : 0.0f;
                const float p1 = ad.y ? __expf(s1) : 0.0f;
                rs += p0 + p1;
                pf.u[r] = pack2bf(p0, p1);
            }
#pragma unroll
            for (int nt = 0; nt < NT; ++nt) {
                union { uint4 q[2]; v16bf v; } bf;
                const uint4* p = (const uint4*)&vl[((jcl * NT) + nt) * 256 + lane * 8];
                bf.q[0] = p[0]; bf.q[1] = p[1];
                acc[nt] = __builtin_amdgcn_wmma_f32_16x16x32_bf16(
                              false, pf.v, false, bf.v, (short)0, acc[nt], false, false);
            }
        }
        __syncthreads();
    }

    // row sums: combine the two K-halves, then broadcast per C-layout row
    const float rst = rs + __shfl_xor(rs, 16, 32);      // row l%16 total
    float inv[8];
#pragma unroll
    for (int r = 0; r < 8; ++r)
        inv[r] = 1.0f / __shfl(rst, 8 * h + r, 32);     // sum of row r+8h

    const int colbase = headcolstride * blockIdx.y;
#pragma unroll
    for (int nt = 0; nt < NT; ++nt) {
        if (nt * 16 + ln < FO) {
            const int c = colbase + nt * 16 + ln;
#pragma unroll
            for (int r = 0; r < 8; ++r) {
                float v = acc[nt][r] * inv[r];
                if (ELU) v = (v > 0.0f) ? v : (__expf(v) - 1.0f);
                out[((size_t)bb * 512 + rowbase + 8 * h + r) * ostride + c] = v;
            }
        }
    }
}

// =====================================================================
extern "C" void kernel_launch(void* const* d_in, const int* in_sizes, int n_in,
                              void* d_out, int out_size, void* d_ws, size_t ws_size,
                              hipStream_t stream) {
    (void)in_sizes; (void)n_in; (void)out_size; (void)ws_size;
    const float* h      = (const float*)d_in[0];   // [32,512,80]
    const int*   adj    = (const int*)  d_in[1];   // [32,512,512]
    const float* Ws     = (const float*)d_in[2];   // [3,80,50]
    const float* attn_a = (const float*)d_in[3];   // [3,100]
    const float* W_out  = (const float*)d_in[4];   // [150,80]
    const float* a_out  = (const float*)d_in[5];   // [160]
    float* out = (float*)d_out;                    // [32,512,80]

    char* ws = (char*)d_ws;
    size_t off = 0;
    auto alloc = [&](size_t bytes) -> void* {
        void* p = ws + off;
        off += (bytes + 255) & ~(size_t)255;
        return p;
    };
    unsigned* vfh = (unsigned*)alloc((size_t)3 * 32 * 16 * 4 * 256 * 4);  // 6.3MB bf16 frags
    float*    e1h = (float*)   alloc((size_t)3 * 32 * 512 * 4);
    float*    e2h = (float*)   alloc((size_t)3 * 32 * 512 * 4);
    float*    cat = (float*)   alloc((size_t)32 * 512 * 160 * 4);         // padded 150->160
    unsigned* vfo = (unsigned*)alloc((size_t)32 * 16 * 5 * 256 * 4);      // 2.6MB
    float*    e1o = (float*)   alloc((size_t)32 * 512 * 4);
    float*    e2o = (float*)   alloc((size_t)32 * 512 * 4);

    // 1) per-head Wh + e1/e2 (f32 WMMA 16x16x4, e folded as extra W columns)
    gat_gemm<20, 80, 4, 50><<<dim3(2, 3, 32), 256, 0, stream>>>(
        h, 80, Ws, attn_a, vfh, e1h, e2h);
    // 2) per-head attention + ELU -> cat[:, :, head*50 : head*50+50]
    gat_attn<4, 50, 1><<<dim3(8, 3, 32), 128, 0, stream>>>(
        vfh, e1h, e2h, adj, cat, 160, 50);
    // 3) output-layer Wh2 = cat @ W_out (K=160, rows >=150 zero-guarded)
    gat_gemm<40, 150, 5, 80><<<dim3(2, 1, 32), 256, 0, stream>>>(
        cat, 160, W_out, a_out, vfo, e1o, e2o);
    // 4) output attention (no ELU) -> d_out
    gat_attn<5, 80, 0><<<dim3(8, 1, 32), 128, 0, stream>>>(
        vfo, e1o, e2o, adj, out, 80, 0);
}
